// Transformer_90812788507167
// MI455X (gfx1250) — matrix operand order repaired and checked by reading, not compiled, not run on hardware
//
#include <hip/hip_runtime.h>
#include <math.h>

// ---------------- types ----------------
typedef __bf16 bf16_t;
typedef __bf16  v16bf __attribute__((ext_vector_type(16)));
typedef float   v8f   __attribute__((ext_vector_type(8)));
typedef int     v4i_t __attribute__((ext_vector_type(4)));

#define BATCH 16
#define SEQ   1024
#define DMODEL 512
#define NHEAD 8
#define DH    64
#define MTOK  (BATCH * SEQ)       // 16384
#define NQKV  (3 * DMODEL)        // 1536

// ---- async global->LDS (CDNA5 GLOBAL_LOAD_ASYNC_TO_LDS_B128, ASYNCcnt) ----
#if defined(__has_builtin)
#if __has_builtin(__builtin_amdgcn_global_load_async_to_lds_b128)
#define HAVE_ASYNC_LDS 1
#endif
#endif
#ifndef HAVE_ASYNC_LDS
#define HAVE_ASYNC_LDS 0
#endif

__device__ __forceinline__ void g2l_b128(const bf16_t* g, bf16_t* l) {
#if HAVE_ASYNC_LDS
    __builtin_amdgcn_global_load_async_to_lds_b128(
        (__attribute__((address_space(1))) v4i_t*)g,
        (__attribute__((address_space(3))) v4i_t*)l, 0, 0);
#else
    *(uint4*)l = *(const uint4*)g;
#endif
}

__device__ __forceinline__ void async_fence() {
#if HAVE_ASYNC_LDS
    asm volatile("s_wait_asynccnt 0x0" ::: "memory");
#endif
}

__device__ __forceinline__ v8f wmma_bf16(v16bf a, v16bf b, v8f c) {
    return __builtin_amdgcn_wmma_f32_16x16x32_bf16(false, a, false, b, (short)0, c, false, false);
}

// ---------------- fp32 -> bf16 convert (4 elems/thread) ----------------
__global__ void cvt_kernel(const float* __restrict__ in, bf16_t* __restrict__ out, int n4) {
    int i = blockIdx.x * blockDim.x + threadIdx.x;
    if (i < n4) {
        float4 t = ((const float4*)in)[i];
        bf16_t* o = out + (size_t)i * 4;
        o[0] = (bf16_t)t.x; o[1] = (bf16_t)t.y; o[2] = (bf16_t)t.z; o[3] = (bf16_t)t.w;
    }
}

// ---------------- LayerNorm (one wave per token, vectorized lanes) ----------------
__global__ __launch_bounds__(128)
void ln_kernel(const float* __restrict__ h, const float* __restrict__ g,
               const float* __restrict__ b, bf16_t* __restrict__ out) {
    const int tok  = blockIdx.x * 4 + (threadIdx.x >> 5);
    const int lane = threadIdx.x & 31;
    const int base = lane * 16;                       // 16 contiguous floats per lane
    const float* row = h + (size_t)tok * DMODEL + base;
    float v[16];
#pragma unroll
    for (int u = 0; u < 4; ++u) {
        float4 t = ((const float4*)row)[u];
        v[u * 4 + 0] = t.x; v[u * 4 + 1] = t.y; v[u * 4 + 2] = t.z; v[u * 4 + 3] = t.w;
    }
    float s = 0.f;
#pragma unroll
    for (int i = 0; i < 16; ++i) s += v[i];
#pragma unroll
    for (int m = 16; m >= 1; m >>= 1) s += __shfl_xor(s, m, 32);
    const float mean = s * (1.0f / DMODEL);
    float vs = 0.f;
#pragma unroll
    for (int i = 0; i < 16; ++i) { float d = v[i] - mean; vs += d * d; }
#pragma unroll
    for (int m = 16; m >= 1; m >>= 1) vs += __shfl_xor(vs, m, 32);
    const float rstd = rsqrtf(vs * (1.0f / DMODEL) + 1e-5f);
    bf16_t* orow = out + (size_t)tok * DMODEL + base;
#pragma unroll
    for (int i = 0; i < 16; ++i)
        orow[i] = (bf16_t)((v[i] - mean) * rstd * g[base + i] + b[base + i]);
}

// ---------------- WMMA GEMM: C[M,N] = A[M,K] * W[N,K]^T + bias ----------------
// Block tile 128(M) x 64(N), K-step 64; 8 waves as 4(M)x2(N), 32x32 per wave.
// LDS tiles filled with GLOBAL_LOAD_ASYNC_TO_LDS_B128 (ASYNCcnt fence + barrier).
// EPI: 0 = store bf16, 1 = exact GELU + store bf16, 2 = residual add fp32
template <int EPI>
__global__ __launch_bounds__(256)
void gemm_kernel(const bf16_t* __restrict__ A, const bf16_t* __restrict__ W,
                 const float* __restrict__ bias, float* __restrict__ resid,
                 bf16_t* __restrict__ outbf, int M, int N, int K) {
    __shared__ __align__(16) bf16_t As[128][72];
    __shared__ __align__(16) bf16_t Bs[64][72];
    const int m0 = blockIdx.y * 128, n0 = blockIdx.x * 64;
    const int tid = threadIdx.x, lane = tid & 31, w = tid >> 5;
    const int wm = w & 3, wn = w >> 2;                // 4x2 wave grid -> 32x32 per wave
    const int ali = tid >> 1;                         // 0..127 A row
    const int alk = (tid & 1) * 32;                   // 0 / 32
    const int bli = tid >> 2;                         // 0..63 B row (output channel)
    const int blk = (tid & 3) * 16;                   // 0,16,32,48

    v8f acc[2][2];
#pragma unroll
    for (int mt = 0; mt < 2; ++mt)
#pragma unroll
        for (int nt = 0; nt < 2; ++nt)
            acc[mt][nt] = (v8f){0.f,0.f,0.f,0.f,0.f,0.f,0.f,0.f};

    const int akb  = (lane < 16) ? 0 : 16;
    const int bkb  = (lane < 16) ? 0 : 16;
    const int ncol = lane & 15;

    for (int kk = 0; kk < K; kk += 64) {
        const bf16_t* ag = A + (size_t)(m0 + ali) * K + kk + alk;
#pragma unroll
        for (int u = 0; u < 4; ++u)
            g2l_b128(ag + u * 8, &As[ali][alk + u * 8]);
        const bf16_t* bg = W + (size_t)(n0 + bli) * K + kk + blk;
#pragma unroll
        for (int u = 0; u < 2; ++u)
            g2l_b128(bg + u * 8, &Bs[bli][blk + u * 8]);
        if (kk + 64 < K) {                            // prefetch next K tile
            __builtin_prefetch(ag + 64, 0, 1);
            __builtin_prefetch(bg + 64, 0, 1);
        }
        async_fence();                                // ASYNCcnt -> 0 before barrier
        __syncthreads();

#pragma unroll
        for (int kc = 0; kc < 2; ++kc) {
            v16bf af[2], bfr[2];
#pragma unroll
            for (int mt = 0; mt < 2; ++mt) {
                const int row = wm * 32 + mt * 16 + ncol;
                const bf16_t* p0 = &As[row][kc * 32 + akb];
                const bf16_t* p1 = &As[row][kc * 32 + 16 + akb];
#pragma unroll
                for (int j = 0; j < 8; ++j) { af[mt][j] = p0[j]; af[mt][8 + j] = p0[8 + j]; }
            }
#pragma unroll
            for (int nt = 0; nt < 2; ++nt) {
                const bf16_t* q = &Bs[wn * 32 + nt * 16 + ncol][kc * 32 + bkb];
#pragma unroll
                for (int j = 0; j < 16; ++j) bfr[nt][j] = q[j];
            }
#pragma unroll
            for (int mt = 0; mt < 2; ++mt)
#pragma unroll
                for (int nt = 0; nt < 2; ++nt)
                    acc[mt][nt] = wmma_bf16(af[mt], bfr[nt], acc[mt][nt]);
        }
        __syncthreads();
    }

    const int roff = (lane >> 4) << 3;
#pragma unroll
    for (int nt = 0; nt < 2; ++nt) {
        const int gn = n0 + wn * 32 + nt * 16 + ncol;
        const float bv = bias[gn];
#pragma unroll
        for (int mt = 0; mt < 2; ++mt) {
            const v8f& a = acc[mt][nt];
#pragma unroll
            for (int g = 0; g < 8; ++g) {
                const int gm = m0 + wm * 32 + mt * 16 + g + roff;
                float val = a[g] + bv;
                if (EPI == 0) {
                    outbf[(size_t)gm * N + gn] = (bf16_t)val;
                } else if (EPI == 1) {
                    float t = 0.5f * val * (1.0f + erff(val * 0.70710678118654752f));
                    outbf[(size_t)gm * N + gn] = (bf16_t)t;
                } else {
                    resid[(size_t)gm * N + gn] += val;
                }
            }
        }
    }
}

// ---------------- Banded windowed attention (one wave per 16-query tile) ----------------
// qkv: [BATCH*SEQ, 1536] bf16 (q|k|v). out: [BATCH*SEQ, 512] bf16.
__global__ __launch_bounds__(32)
void attn_kernel(const bf16_t* __restrict__ qkv, const float* __restrict__ mask,
                 bf16_t* __restrict__ out) {
    const int qt = blockIdx.x & 63;
    const int h  = (blockIdx.x >> 6) & 7;
    const int b  = blockIdx.x >> 9;
    const int lane = threadIdx.x;
    const int q0 = qt * 16;
    const int kt_lo = (q0 >= 33) ? ((q0 - 33) >> 4) : 0;
    const int kt_hi_raw = (q0 + 48) >> 4;
    const int kt_hi = kt_hi_raw > 63 ? 63 : kt_hi_raw;
    const int nkt = kt_hi - kt_lo + 1;            // 4..7 valid key tiles

    const size_t tokbase = (size_t)b * SEQ;
    const bf16_t* Qp = qkv + (tokbase + q0) * NQKV + h * DH;
    const bf16_t* Kp = qkv + tokbase * NQKV + DMODEL + h * DH;
    const bf16_t* Vb = qkv + (tokbase + kt_lo * 16) * NQKV + 2 * DMODEL + h * DH;

    const int arow = lane & 15;
    const int akb  = (lane < 16) ? 0 : 16;
    const int bkb  = (lane < 16) ? 0 : 16;
    const int ncol = lane & 15;
    const int roff = (lane >> 4) << 3;

    // ---- stage banded V tile transposed into LDS: Vt[d][key], zero-padded ----
    __shared__ __align__(16) bf16_t Vt[64][136];
    {
        const int vkey = lane >> 3;               // 0..3
        const int vd0  = (lane & 7) * 8;          // 0..56
        const int nkeys = nkt * 16;
#pragma unroll
        for (int pass = 0; pass < 32; ++pass) {
            const int key = pass * 4 + vkey;      // 0..127
            if (key < nkeys) {
                const uint4 t = *(const uint4*)(Vb + (size_t)key * NQKV + vd0);
                const bf16_t* e = (const bf16_t*)&t;
#pragma unroll
                for (int i = 0; i < 8; ++i) Vt[vd0 + i][key] = e[i];
            } else {
#pragma unroll
                for (int i = 0; i < 8; ++i) Vt[vd0 + i][key] = (bf16_t)0.0f;
            }
        }
    }

    // ---- Q fragments (A layout) for the two K=32 steps covering dh=64 ----
    v16bf qa[2];
#pragma unroll
    for (int s = 0; s < 2; ++s) {
        const bf16_t* r = Qp + (size_t)arow * NQKV + s * 32;
#pragma unroll
        for (int j = 0; j < 8; ++j) { qa[s][j] = r[akb + j]; qa[s][8 + j] = r[8 + akb + j]; }
    }

    // ---- Pass 1: raw scores over the band, track per-row max ----
    float sc[7][8];
    float mx[8];
#pragma unroll
    for (int g = 0; g < 8; ++g) mx[g] = -3.0e38f;

#pragma unroll
    for (int t = 0; t < 7; ++t) {
        if (t < nkt) {
            const int k0 = (kt_lo + t) * 16;
            v8f a = {0.f,0.f,0.f,0.f,0.f,0.f,0.f,0.f};
#pragma unroll
            for (int s = 0; s < 2; ++s) {
                v16bf kf;
                const bf16_t* r = Kp + (size_t)(k0 + ncol) * NQKV + s * 32 + bkb;
#pragma unroll
                for (int j = 0; j < 16; ++j) kf[j] = r[j];
                a = wmma_bf16(qa[s], kf, a);
            }
#pragma unroll
            for (int g = 0; g < 8; ++g) {
                float v = a[g] * 0.125f + mask[(size_t)(q0 + g + roff) * SEQ + k0 + ncol];
                sc[t][g] = v;
                mx[g] = fmaxf(mx[g], v);
            }
        } else {
#pragma unroll
            for (int g = 0; g < 8; ++g) sc[t][g] = -3.0e38f;
        }
    }
#pragma unroll
    for (int g = 0; g < 8; ++g)
#pragma unroll
        for (int m = 1; m <= 8; m <<= 1)
            mx[g] = fmaxf(mx[g], __shfl_xor(mx[g], m, 32));

    // ---- Pass 2: exp, park P (unnormalized) in LDS, accumulate row sums ----
    __shared__ __align__(16) bf16_t Pl[16][136];   // 128 key slots + pad
    float rs[8];
#pragma unroll
    for (int g = 0; g < 8; ++g) rs[g] = 0.f;
#pragma unroll
    for (int t = 0; t < 7; ++t) {
#pragma unroll
        for (int g = 0; g < 8; ++g) {
            float e = __expf(sc[t][g] - mx[g]);
            rs[g] += e;
            Pl[g + roff][t * 16 + ncol] = (bf16_t)e;
        }
    }
#pragma unroll
    for (int g = 0; g < 8; ++g) Pl[g + roff][112 + ncol] = (bf16_t)0.0f;  // zero slot 7
#pragma unroll
    for (int g = 0; g < 8; ++g)
#pragma unroll
        for (int m = 1; m <= 8; m <<= 1)
            rs[g] += __shfl_xor(rs[g], m, 32);
    __syncthreads();

    // ---- O = P @ V over 4 key pairs (K=32 each) x 4 d-tiles ----
    v8f o[4];
#pragma unroll
    for (int t = 0; t < 4; ++t) o[t] = (v8f){0.f,0.f,0.f,0.f,0.f,0.f,0.f,0.f};

#pragma unroll
    for (int p = 0; p < 4; ++p) {
        v16bf pa;
        const bf16_t* r = &Pl[arow][p * 32];
#pragma unroll
        for (int j = 0; j < 8; ++j) { pa[j] = r[akb + j]; pa[8 + j] = r[8 + akb + j]; }
#pragma unroll
        for (int t4 = 0; t4 < 4; ++t4) {
            v16bf vf;
            const bf16_t* q = &Vt[t4 * 16 + ncol][p * 32 + bkb];
#pragma unroll
            for (int j = 0; j < 16; ++j) vf[j] = q[j];
            o[t4] = wmma_bf16(pa, vf, o[t4]);
        }
    }

    // ---- normalize and store ----
#pragma unroll
    for (int t4 = 0; t4 < 4; ++t4) {
#pragma unroll
        for (int g = 0; g < 8; ++g) {
            const int row = g + roff;
            out[(tokbase + q0 + row) * DMODEL + h * DH + t4 * 16 + ncol] =
                (bf16_t)(o[t4][g] / rs[g]);
        }
    }
}

// ---------------- host orchestration ----------------
extern "C" void kernel_launch(void* const* d_in, const int* in_sizes, int n_in,
                              void* d_out, int out_size, void* d_ws, size_t ws_size,
                              hipStream_t stream) {
    const float* x     = (const float*)d_in[0];
    const float* ln1_g = (const float*)d_in[1];
    const float* ln1_b = (const float*)d_in[2];
    const float* wqkv  = (const float*)d_in[3];
    const float* bqkv  = (const float*)d_in[4];
    const float* wo    = (const float*)d_in[5];
    const float* bo    = (const float*)d_in[6];
    const float* ln2_g = (const float*)d_in[7];
    const float* ln2_b = (const float*)d_in[8];
    const float* w1    = (const float*)d_in[9];
    const float* b1    = (const float*)d_in[10];
    const float* w2    = (const float*)d_in[11];
    const float* b2    = (const float*)d_in[12];
    const float* mask  = (const float*)d_in[13];
    float* h = (float*)d_out;   // running residual stream, fp32

    // workspace layout (bf16)
    bf16_t* ws    = (bf16_t*)d_ws;
    bf16_t* wq_bf = ws;                               // 1536*512
    bf16_t* wo_bf = wq_bf + NQKV * DMODEL;            // 512*512
    bf16_t* w1_bf = wo_bf + DMODEL * DMODEL;          // 512*512
    bf16_t* w2_bf = w1_bf + DMODEL * DMODEL;          // 512*512
    bf16_t* nbuf  = w2_bf + DMODEL * DMODEL;          // 16384*512
    bf16_t* qkvb  = nbuf + (size_t)MTOK * DMODEL;     // 16384*1536
    bf16_t* obuf  = qkvb + (size_t)MTOK * NQKV;       // 16384*512 (also FF-gelu buf)

    // seed residual stream h = x
    (void)hipMemcpyAsync(h, x, (size_t)MTOK * DMODEL * sizeof(float),
                         hipMemcpyDeviceToDevice, stream);

    const dim3 blk256(256), blk128(128), blk32(32);
    const dim3 g_qkv(NQKV / 64, MTOK / 128);
    const dim3 g_sq(DMODEL / 64, MTOK / 128);

    for (int l = 0; l < 6; ++l) {
        // convert this layer's weights to bf16 (4 elems/thread)
        cvt_kernel<<<(NQKV * DMODEL / 4) / 256, blk256, 0, stream>>>(
            wqkv + (size_t)l * NQKV * DMODEL, wq_bf, NQKV * DMODEL / 4);
        cvt_kernel<<<(DMODEL * DMODEL / 4) / 256, blk256, 0, stream>>>(
            wo + (size_t)l * DMODEL * DMODEL, wo_bf, DMODEL * DMODEL / 4);
        cvt_kernel<<<(DMODEL * DMODEL / 4) / 256, blk256, 0, stream>>>(
            w1 + (size_t)l * DMODEL * DMODEL, w1_bf, DMODEL * DMODEL / 4);
        cvt_kernel<<<(DMODEL * DMODEL / 4) / 256, blk256, 0, stream>>>(
            w2 + (size_t)l * DMODEL * DMODEL, w2_bf, DMODEL * DMODEL / 4);

        // LN1
        ln_kernel<<<MTOK / 4, blk128, 0, stream>>>(
            h, ln1_g + l * DMODEL, ln1_b + l * DMODEL, nbuf);
        // QKV projection -> bf16
        gemm_kernel<0><<<g_qkv, blk256, 0, stream>>>(
            nbuf, wq_bf, bqkv + (size_t)l * NQKV, nullptr, qkvb, MTOK, NQKV, DMODEL);
        // windowed attention
        attn_kernel<<<BATCH * NHEAD * (SEQ / 16), blk32, 0, stream>>>(qkvb, mask, obuf);
        // output projection + residual into h
        gemm_kernel<2><<<g_sq, blk256, 0, stream>>>(
            obuf, wo_bf, bo + (size_t)l * DMODEL, h, nullptr, MTOK, DMODEL, DMODEL);
        // LN2
        ln_kernel<<<MTOK / 4, blk128, 0, stream>>>(
            h, ln2_g + l * DMODEL, ln2_b + l * DMODEL, nbuf);
        // FF1 + GELU -> bf16 (reuse obuf)
        gemm_kernel<1><<<g_sq, blk256, 0, stream>>>(
            nbuf, w1_bf, b1 + (size_t)l * DMODEL, nullptr, obuf, MTOK, DMODEL, DMODEL);
        // FF2 + residual into h
        gemm_kernel<2><<<g_sq, blk256, 0, stream>>>(
            obuf, w2_bf, b2 + (size_t)l * DMODEL, h, nullptr, MTOK, DMODEL, DMODEL);
    }
}
